// MultiHeadAttention_57251914056150
// MI455X (gfx1250) — matrix operand order repaired and checked by reading, not compiled, not run on hardware
//
#include <hip/hip_runtime.h>
#include <math.h>

// MI455X / gfx1250: wave32, WMMA bf16 path (v_wmma_f32_16x16x32_bf16),
// async global->LDS staging (ASYNCcnt) with double buffering where available.
typedef __attribute__((ext_vector_type(8)))  __bf16 v8bf;
typedef __attribute__((ext_vector_type(16))) __bf16 v16bf;
typedef __attribute__((ext_vector_type(8)))  float  v8f;
typedef __attribute__((ext_vector_type(4)))  int    v4i;

#define B_  4
#define S_  2048
#define E_  1024
#define H_  16
#define HD_ 64

#if __has_builtin(__builtin_amdgcn_global_load_async_to_lds_b128)
#define HAS_ASYNC_LDS 1
#endif

__device__ inline void cp16_async(const __bf16* g, __bf16* l) {
#if defined(HAS_ASYNC_LDS)
  // signature (from hipcc diagnostic): (global v4i*, local v4i*, imm offset, imm cpol)
  __builtin_amdgcn_global_load_async_to_lds_b128(
      (__attribute__((address_space(1))) v4i*)g,
      (__attribute__((address_space(3))) v4i*)l, 0, 0);
#else
  *(v8bf*)l = *(const v8bf*)g;
#endif
}

__device__ inline void wait_async0() {
#if defined(HAS_ASYNC_LDS)
#if __has_builtin(__builtin_amdgcn_s_wait_asynccnt)
  __builtin_amdgcn_s_wait_asynccnt(0);
#else
  asm volatile("s_wait_asynccnt 0x0" ::: "memory");
#endif
#endif
}

__device__ inline v8f wmma_bf16(v16bf a, v16bf b, v8f c) {
  return __builtin_amdgcn_wmma_f32_16x16x32_bf16(false, a, false, b, (short)0, c,
                                                 false, false);
}

__device__ inline v16bf cat8(v8bf lo, v8bf hi) {
  return __builtin_shufflevector(lo, hi, 0, 1, 2, 3, 4, 5, 6, 7,
                                 8, 9, 10, 11, 12, 13, 14, 15);
}

// A fragment 16x32 bf16: lane (m=lane&15, hf=lane>>4) reads two contiguous
// 16B chunks at K = 8*hf and K = 16+8*hf (ISA 7.12.2).  ld must be mult of 8.
__device__ inline v16bf ld_frag_a(const __bf16* t, int ld) {
  int lane = threadIdx.x & 31, m = lane & 15, hf = lane >> 4;
  const __bf16* p = t + m * ld + 8 * hf;
  return cat8(*(const v8bf*)p, *(const v8bf*)(p + 16));
}
// B fragment 32x16 bf16 from a K-major tile (row = N, K contiguous):
// lane (n, hf) reads two 16B chunks at row n, K = 8*hf and K = 16+8*hf.
__device__ inline v16bf ld_frag_bk(const __bf16* t, int ld) {
  int lane = threadIdx.x & 31, n = lane & 15, hf = lane >> 4;
  const __bf16* p = t + n * ld + 8 * hf;
  return cat8(*(const v8bf*)p, *(const v8bf*)(p + 16));
}

// ============ 0) one-time: fp32 -> bf16 transpose of weight matrices =======
__global__ void transpose_cvt_kernel(const float* __restrict__ src,
                                     __bf16* __restrict__ dst, int R, int C) {
  long m = blockIdx.y;
  int i = blockIdx.x * 256 + threadIdx.x;
  if (i < R * C) {
    int r = i / C, c = i % C;
    dst[m * R * C + (long)c * R + r] = (__bf16)src[m * R * C + i];
  }
}

// ======================= 1) per-head QKV projection ========================
__global__ __launch_bounds__(128) void qkv_proj_kernel(
    const float* __restrict__ x, const __bf16* __restrict__ WT,
    const float* __restrict__ bias, __bf16* __restrict__ out, int transposed) {
  __shared__ __bf16 WtT[64 * 72];  // [e][d]
  __shared__ __bf16 Xt[64 * 72];   // [row][d]
  int h = blockIdx.y;
  int t = threadIdx.x;
  long grow0 = (long)blockIdx.x * 64;  // row block over B*S

  const __bf16* Wh = WT + (long)h * 64 * 64;
#pragma unroll
  for (int i = 0; i < 4; ++i) {
    int e8 = (t + i * 128) * 8;  // 0..4095
    int r = e8 >> 6, c = e8 & 63;
    cp16_async(&Wh[e8], &WtT[r * 72 + c]);
    const float4 f0 = *(const float4*)&x[(grow0 + r) * E_ + h * 64 + c];
    const float4 f1 = *(const float4*)&x[(grow0 + r) * E_ + h * 64 + c + 4];
    v8bf pk;
    pk[0] = (__bf16)f0.x; pk[1] = (__bf16)f0.y; pk[2] = (__bf16)f0.z;
    pk[3] = (__bf16)f0.w; pk[4] = (__bf16)f1.x; pk[5] = (__bf16)f1.y;
    pk[6] = (__bf16)f1.z; pk[7] = (__bf16)f1.w;
    *(v8bf*)&Xt[r * 72 + c] = pk;
  }
  wait_async0();
  __syncthreads();

  int w = t >> 5, lane = t & 31;
  int n = lane & 15, hf = lane >> 4;
  v16bf a0 = ld_frag_a(Xt + w * 16 * 72, 72);
  v16bf a1 = ld_frag_a(Xt + w * 16 * 72 + 32, 72);

  long b = grow0 / S_;
  long bh = b * H_ + h;
  long s0 = grow0 % S_ + w * 16;
#pragma unroll
  for (int nt = 0; nt < 4; ++nt) {
    v8f c = {};
    c = wmma_bf16(a0, ld_frag_bk(WtT + nt * 16 * 72, 72), c);
    c = wmma_bf16(a1, ld_frag_bk(WtT + nt * 16 * 72 + 32, 72), c);
    float bcol = bias[h * 64 + nt * 16 + n];
    if (transposed) {  // V: [B,H,HD,S]; lane's 8 values are consecutive s
      v8bf pk;
#pragma unroll
      for (int j = 0; j < 8; ++j) pk[j] = (__bf16)(c[j] + bcol);
      *(v8bf*)&out[(bh * HD_ + nt * 16 + n) * S_ + s0 + 8 * hf] = pk;
    } else {           // Q/K: [B,H,S,HD]
#pragma unroll
      for (int v = 0; v < 8; ++v)
        out[(bh * S_ + s0 + v + 8 * hf) * HD_ + nt * 16 + n] =
            (__bf16)(c[v] + bcol);
    }
  }
}

// ============================ 2) flash attention ===========================
__global__ __launch_bounds__(128) void flash_attn_kernel(
    const __bf16* __restrict__ Q,   // [B,H,S,HD]
    const __bf16* __restrict__ K,   // [B,H,S,HD]
    const __bf16* __restrict__ VT,  // [B,H,HD,S]
    __bf16* __restrict__ ctx,       // [B,S,E]
    const int* __restrict__ maskp) {
  __shared__ __bf16 Kt[2][32 * 72];    // [key][hd]  (K-major for Q.K^T)
  __shared__ __bf16 VtT[2][64 * 40];   // [hd][key]  (K-major for P.V)
  __shared__ __bf16 Pt[4][16 * 40];    // per-wave P transpose scratch

  int bh = blockIdx.y;
  int b = bh / H_, h = bh % H_;
  int q0wg = blockIdx.x * 64;
  int t = threadIdx.x, w = t >> 5, lane = t & 31;
  int n = lane & 15, hf = lane >> 4;
  int q0 = q0wg + w * 16;
  int causal = *maskp;

  const __bf16* Qb = Q + (long)bh * S_ * HD_;
  const __bf16* Kb = K + (long)bh * S_ * HD_;
  const __bf16* Vb = VT + (long)bh * HD_ * S_;

  // Q fragments: 4 x global_load_b128 per lane
  const __bf16* qrow = Qb + (q0 + (lane & 15)) * HD_;
  v16bf qf0 = cat8(*(const v8bf*)(qrow + 8 * hf), *(const v8bf*)(qrow + 16 + 8 * hf));
  v16bf qf1 = cat8(*(const v8bf*)(qrow + 32 + 8 * hf), *(const v8bf*)(qrow + 48 + 8 * hf));

  v8f acc0 = {}, acc1 = {}, acc2 = {}, acc3 = {};
  float mrow[8], lrow[8];
#pragma unroll
  for (int v = 0; v < 8; ++v) { mrow[v] = -1e30f; lrow[v] = 0.f; }

  int kmax = causal ? (q0wg + 64) : S_;
  int nblocks = (kmax + 31) / 32;

  auto stage = [&](int kb, int bufi) {  // async prefetch of one 32-key block
    int k0 = kb * 32;
    int r = t >> 2, c = (t & 3) * 16;
    cp16_async(&Kb[(long)(k0 + r) * HD_ + c],     &Kt[bufi][r * 72 + c]);
    cp16_async(&Kb[(long)(k0 + r) * HD_ + c + 8], &Kt[bufi][r * 72 + c + 8]);
    int hd = t >> 1, ck = (t & 1) * 16;
    cp16_async(&Vb[(long)hd * S_ + k0 + ck],     &VtT[bufi][hd * 40 + ck]);
    cp16_async(&Vb[(long)hd * S_ + k0 + ck + 8], &VtT[bufi][hd * 40 + ck + 8]);
  };

  stage(0, 0);
  wait_async0();
  __syncthreads();

  for (int kb = 0; kb < nblocks; ++kb) {
    int cur = kb & 1;
    int k0 = kb * 32;
    if (kb + 1 < nblocks) stage(kb + 1, cur ^ 1);  // overlap with compute

    bool active = (!causal) || (k0 <= q0 + 15);
    if (active) {
      const __bf16* Kc = &Kt[cur][0];
      const __bf16* Vc = &VtT[cur][0];
      // interleave the two independent score chains (hides WMMA RAW hazard)
      v8f s0 = {}, s1 = {};
      s0 = wmma_bf16(qf0, ld_frag_bk(Kc, 72), s0);
      s1 = wmma_bf16(qf0, ld_frag_bk(Kc + 16 * 72, 72), s1);
      s0 = wmma_bf16(qf1, ld_frag_bk(Kc + 32, 72), s0);
      s1 = wmma_bf16(qf1, ld_frag_bk(Kc + 16 * 72 + 32, 72), s1);

#pragma unroll
      for (int v = 0; v < 8; ++v) { s0[v] *= 0.125f; s1[v] *= 0.125f; }
      if (causal) {
#pragma unroll
        for (int v = 0; v < 8; ++v) {
          int qr = q0 + v + 8 * hf;
          if (k0 + n > qr)      s0[v] = -1e30f;
          if (k0 + 16 + n > qr) s1[v] = -1e30f;
        }
      }
#pragma unroll
      for (int v = 0; v < 8; ++v) {
        float mx = fmaxf(s0[v], s1[v]);
        mx = fmaxf(mx, __shfl_xor(mx, 1));
        mx = fmaxf(mx, __shfl_xor(mx, 2));
        mx = fmaxf(mx, __shfl_xor(mx, 4));
        mx = fmaxf(mx, __shfl_xor(mx, 8));
        float mnew = fmaxf(mrow[v], mx);
        float alpha = __expf(mrow[v] - mnew);
        float p0 = __expf(s0[v] - mnew);
        float p1 = __expf(s1[v] - mnew);
        float r = p0 + p1;
        r += __shfl_xor(r, 1); r += __shfl_xor(r, 2);
        r += __shfl_xor(r, 4); r += __shfl_xor(r, 8);
        lrow[v] = lrow[v] * alpha + r;
        mrow[v] = mnew;
        acc0[v] *= alpha; acc1[v] *= alpha; acc2[v] *= alpha; acc3[v] *= alpha;
        s0[v] = p0; s1[v] = p1;
      }
      // P: C layout -> A layout via per-wave LDS scratch
      __bf16* Pw = &Pt[w][0];
#pragma unroll
      for (int v = 0; v < 8; ++v) {
        int m = v + 8 * hf;
        Pw[m * 40 + n]      = (__bf16)s0[v];
        Pw[m * 40 + 16 + n] = (__bf16)s1[v];
      }
      asm volatile("s_wait_dscnt 0x0" ::: "memory");  // intra-wave cross-lane RAW
      v16bf pf = ld_frag_a(Pw, 40);
      acc0 = wmma_bf16(pf, ld_frag_bk(Vc + 0 * 16 * 40, 40), acc0);
      acc1 = wmma_bf16(pf, ld_frag_bk(Vc + 1 * 16 * 40, 40), acc1);
      acc2 = wmma_bf16(pf, ld_frag_bk(Vc + 2 * 16 * 40, 40), acc2);
      acc3 = wmma_bf16(pf, ld_frag_bk(Vc + 3 * 16 * 40, 40), acc3);
    }
    wait_async0();    // our prefetch writes have landed in LDS
    __syncthreads();  // publish buffer to all waves / retire reads
  }

#pragma unroll
  for (int v = 0; v < 8; ++v) {
    float inv = 1.0f / lrow[v];
    long row = (long)b * S_ + q0 + v + 8 * hf;
    __bf16* dst = ctx + row * E_ + h * 64 + n;
    dst[0]  = (__bf16)(acc0[v] * inv);
    dst[16] = (__bf16)(acc1[v] * inv);
    dst[32] = (__bf16)(acc2[v] * inv);
    dst[48] = (__bf16)(acc3[v] * inv);
  }
}

// ===================== 3) output projection (GEMM) =========================
// workgroup tile 128x64, wave tile 32x64, K-step 32, double-buffered async.
__global__ __launch_bounds__(128) void out_proj_kernel(
    const __bf16* __restrict__ A, const __bf16* __restrict__ WpT,
    const float* __restrict__ bias, float* __restrict__ out) {
  __shared__ __bf16 At[2][128 * 40];   // [row][k]
  __shared__ __bf16 BtT[2][64 * 40];   // [n][k]
  int m0 = blockIdx.x * 128;
  int n0 = blockIdx.y * 64;
  int t = threadIdx.x, w = t >> 5, lane = t & 31;
  int n = lane & 15, hf = lane >> 4;
  v8f cA[4] = {{}, {}, {}, {}};
  v8f cB[4] = {{}, {}, {}, {}};

  auto stage = [&](int kk, int bufi) {
#pragma unroll
    for (int i = 0; i < 4; ++i)  // A: 128 rows x 32 k, one row per thread
      cp16_async(&A[(long)(m0 + t) * E_ + kk + 8 * i], &At[bufi][t * 40 + 8 * i]);
    int r = t >> 1, ck = (t & 1) * 16;  // B: 64 n-rows x 32 k
    cp16_async(&WpT[(long)(n0 + r) * E_ + kk + ck],     &BtT[bufi][r * 40 + ck]);
    cp16_async(&WpT[(long)(n0 + r) * E_ + kk + ck + 8], &BtT[bufi][r * 40 + ck + 8]);
  };

  stage(0, 0);
  wait_async0();
  __syncthreads();

  for (int it = 0; it < E_ / 32; ++it) {
    int cur = it & 1;
    if (it + 1 < E_ / 32) stage((it + 1) * 32, cur ^ 1);

    v16bf a0 = ld_frag_a(&At[cur][(w * 32) * 40], 40);
    v16bf a1 = ld_frag_a(&At[cur][(w * 32 + 16) * 40], 40);
#pragma unroll
    for (int nt = 0; nt < 4; ++nt) {
      v16bf bfrag = ld_frag_bk(&BtT[cur][nt * 16 * 40], 40);
      cA[nt] = wmma_bf16(a0, bfrag, cA[nt]);
      cB[nt] = wmma_bf16(a1, bfrag, cB[nt]);
    }
    wait_async0();
    __syncthreads();
  }
#pragma unroll
  for (int nt = 0; nt < 4; ++nt) {
    float bcol = bias[n0 + nt * 16 + n];
#pragma unroll
    for (int v = 0; v < 8; ++v) {
      int m = w * 32 + v + 8 * hf;
      out[(long)(m0 + m) * E_ + n0 + nt * 16 + n] = cA[nt][v] + bcol;
      out[(long)(m0 + m + 16) * E_ + n0 + nt * 16 + n] = cB[nt][v] + bcol;
    }
  }
}

extern "C" void kernel_launch(void* const* d_in, const int* in_sizes, int n_in,
                              void* d_out, int out_size, void* d_ws, size_t ws_size,
                              hipStream_t stream) {
  (void)in_sizes; (void)n_in; (void)out_size; (void)ws_size;
  const float* q  = (const float*)d_in[0];
  const float* k  = (const float*)d_in[1];
  const float* v  = (const float*)d_in[2];
  const float* Wq = (const float*)d_in[3];
  const float* bq = (const float*)d_in[4];
  const float* Wk = (const float*)d_in[5];
  const float* bk = (const float*)d_in[6];
  const float* Wv = (const float*)d_in[7];
  const float* bv = (const float*)d_in[8];
  const float* Wp = (const float*)d_in[9];
  const float* bp = (const float*)d_in[10];
  const int* mask = (const int*)d_in[11];
  float* out = (float*)d_out;

  const size_t QKV = (size_t)B_ * H_ * S_ * HD_;
  __bf16* Qb  = (__bf16*)d_ws;              // [B,H,S,HD]
  __bf16* Kb  = Qb + QKV;                   // [B,H,S,HD]
  __bf16* VbT = Kb + QKV;                   // [B,H,HD,S]
  __bf16* Cb  = VbT + QKV;                  // ctx [B,S,E]
  __bf16* WpT = Cb + (size_t)B_ * S_ * E_;  // [N][K] bf16
  __bf16* WqT = WpT + (size_t)E_ * E_;      // [H][e][d] bf16
  __bf16* WkT = WqT + (size_t)H_ * 64 * 64;
  __bf16* WvT = WkT + (size_t)H_ * 64 * 64;

  transpose_cvt_kernel<<<dim3(16, H_), 256, 0, stream>>>(Wq, WqT, 64, 64);
  transpose_cvt_kernel<<<dim3(16, H_), 256, 0, stream>>>(Wk, WkT, 64, 64);
  transpose_cvt_kernel<<<dim3(16, H_), 256, 0, stream>>>(Wv, WvT, 64, 64);
  transpose_cvt_kernel<<<dim3(4096, 1), 256, 0, stream>>>(Wp, WpT, E_, E_);

  dim3 gA(B_ * S_ / 64, H_);
  qkv_proj_kernel<<<gA, 128, 0, stream>>>(q, WqT, bq, Qb, 0);
  qkv_proj_kernel<<<gA, 128, 0, stream>>>(k, WkT, bk, Kb, 0);
  qkv_proj_kernel<<<gA, 128, 0, stream>>>(v, WvT, bv, VbT, 1);

  dim3 gF(S_ / 64, B_ * H_);
  flash_attn_kernel<<<gF, 128, 0, stream>>>(Qb, Kb, VbT, Cb, mask);

  dim3 gP(B_ * S_ / 128, E_ / 64);
  out_proj_kernel<<<gP, 128, 0, stream>>>(Cb, WpT, bp, out);
}
